// EdgeDegreeEmbeddingNetwork_higherorder_89979564851620
// MI455X (gfx1250) — matrix operand order repaired and checked by reading, not compiled, not run on hardware
//
#include <hip/hip_runtime.h>
#include <hip/hip_bf16.h>

// EdgeDegreeEmbeddingNetwork_higherorder for MI455X (gfx1250, wave32, WMMA).
// N=10000 atoms, K=32 neighbors, NB=64, C=128, L=3, H=64.
// Prep kernel swizzles fp32 weights -> f16 wave32 B-fragments in d_ws.
// Main kernel: one WG (8 wave32) per atom; GEMMs via v_wmma_f32_16x16x32_f16,
// LN/SiLU via wave32 shuffles, neighbor einsum fused out of WMMA accumulators
// into LDS via ds_add_f32.

#define KNB 32      // neighbors
#define NB  64      // basis
#define CH  128     // channels
#define HID 64      // hidden
#define DIN 192     // 3*NB
#define NL  3       // l orders
#define LN_EPS 1e-5f

// fragment table in d_ws (halfs): 512 halfs per fragment (32 lanes x 16)
#define NFRAG_W1 72            // 3l * 4nt * 6ks
#define NFRAG_W2 24            // 3l * 4nt * 2ks
#define NFRAG_W3 48            // 3l * 8ct * 2ks
#define NFRAG    (NFRAG_W1 + NFRAG_W2 + NFRAG_W3)   // 144
#define WS_W2_OFF (NFRAG_W1 * 512)                  // 36864 halfs
#define WS_W3_OFF ((NFRAG_W1 + NFRAG_W2) * 512)     // 49152 halfs

typedef __attribute__((ext_vector_type(16))) _Float16 v16h;
typedef __attribute__((ext_vector_type(8)))  _Float16 v8h;
typedef __attribute__((ext_vector_type(8)))  float    v8f;

__device__ __forceinline__ v8f wmma_f16(v16h a, v16h b, v8f c) {
    // D = A(16x32 f16) * B(32x16 f16) + C(16x16 f32)
    return __builtin_amdgcn_wmma_f32_16x16x32_f16(
        false, a, false, b, (short)0, c, false, false);
}

// A fragment (16x32 f16) from row-major LDS tile; two aligned 16B vector loads.
// Lane L: row m0+(L&15); halves 0..7 -> K=k0+(L>=16?8:0)+h, halves 8..15 -> +16.
__device__ __forceinline__ v16h load_A_lds(const _Float16* X, int m0, int k0, int ld) {
    const int lane = threadIdx.x & 31;
    const _Float16* p = X + (m0 + (lane & 15)) * ld + k0 + ((lane >> 4) << 3);
    v8h lo = *(const v8h*)p;          // 16B aligned (k0 mult of 32, offset 0/16)
    v8h hi = *(const v8h*)(p + 16);
    return __builtin_shufflevector(lo, hi, 0, 1, 2, 3, 4, 5, 6, 7,
                                           8, 9, 10, 11, 12, 13, 14, 15);
}

// B fragment: pre-swizzled in ws -> one 32B vector load per lane.
__device__ __forceinline__ v16h load_B_ws(const _Float16* __restrict__ frags, int fragIdx) {
    const int lane = threadIdx.x & 31;
    return *(const v16h*)(frags + (((size_t)fragIdx << 5) + lane) * 16);
}

// C/D fragment element i <-> (row m0 + i + 8*(lane>>4), col n0 + (lane&15)).
__device__ __forceinline__ void store_acc(float* H, int m0, int n0, int ld, v8f acc) {
    const int lane = threadIdx.x & 31;
    const int c  = n0 + (lane & 15);
    const int mb = m0 + ((lane >> 4) << 3);
#pragma unroll
    for (int i = 0; i < 8; ++i) H[(mb + i) * ld + c] = acc[i];
}

__device__ __forceinline__ float wave_sum32(float v) {
#pragma unroll
    for (int off = 16; off > 0; off >>= 1) v += __shfl_xor(v, off, 32);
    return v;
}

// LayerNorm (+pre-bias) + SiLU over 96 rows of 64, wave-per-row, f32->f16.
__device__ __forceinline__ void ln_silu(const float* __restrict__ bvec,
                                        const float* __restrict__ g,
                                        const float* __restrict__ be,
                                        const float* sHraw, _Float16* sH16) {
    const int wave = threadIdx.x >> 5, lane = threadIdx.x & 31;
    for (int row = wave; row < NL * KNB; row += 8) {
        const int l = row >> 5;
        const float* src = sHraw + row * HID;
        float x0 = src[lane]      + bvec[l * HID + lane];
        float x1 = src[lane + 32] + bvec[l * HID + lane + 32];
        float mu = wave_sum32(x0 + x1) * (1.0f / 64.0f);
        float d0 = x0 - mu, d1 = x1 - mu;
        float var = wave_sum32(d0 * d0 + d1 * d1) * (1.0f / 64.0f);
        float inv = rsqrtf(var + LN_EPS);
        float y0 = d0 * inv * g[l * HID + lane]      + be[l * HID + lane];
        float y1 = d1 * inv * g[l * HID + lane + 32] + be[l * HID + lane + 32];
        y0 = y0 / (1.0f + __expf(-y0));   // SiLU
        y1 = y1 / (1.0f + __expf(-y1));
        _Float16* dst = sH16 + row * HID;
        dst[lane]      = (_Float16)y0;
        dst[lane + 32] = (_Float16)y1;
    }
}

template <int ND, int DB>
__device__ __forceinline__ void reduce_einsum(v8f acc, int mt, int ct,
                                              const float* __restrict__ b3l,
                                              const float* sSh, float* sOut) {
    const int lane = threadIdx.x & 31;
    const int c  = ct * 16 + (lane & 15);
    const int mb = mt * 16 + ((lane >> 4) << 3);
    const float bias = b3l[c];
    float ov[ND];
#pragma unroll
    for (int d = 0; d < ND; ++d) ov[d] = 0.0f;
#pragma unroll
    for (int i = 0; i < 8; ++i) {
        const float v = acc[i] + bias;                 // h = h@W3 + b3
        const float* shr = sSh + (mb + i) * 9 + DB;    // mask folded into sh
#pragma unroll
        for (int d = 0; d < ND; ++d) ov[d] += shr[d] * v;
    }
#pragma unroll
    for (int d = 0; d < ND; ++d) {
        float tot = ov[d] + __shfl_xor(ov[d], 16, 32); // combine M halves (same c)
        if (lane < 16) atomicAdd(&sOut[(DB + d) * CH + c], tot); // ds_add_f32
    }
}

// ---------------- prep: fp32 weights -> f16 wave32 B-fragments in ws ----------
__global__ __launch_bounds__(256)
void edn_prep_weights_kernel(const float* __restrict__ W1,
                             const float* __restrict__ W2,
                             const float* __restrict__ W3,
                             _Float16* __restrict__ ws) {
    const int gt = blockIdx.x * 256 + threadIdx.x;
    if (gt >= NFRAG * 32) return;
    const int frag = gt >> 5, lane = gt & 31;
    const float* src; int ld, k0, n0;
    if (frag < NFRAG_W1) {                       // W1[l] (192x64): 6 ksteps, 4 ntiles
        const int ks = frag % 6, ln = frag / 6;
        src = W1 + (ln >> 2) * DIN * HID; ld = HID; k0 = ks * 32; n0 = (ln & 3) * 16;
    } else if (frag < NFRAG_W1 + NFRAG_W2) {     // W2[l] (64x64): 2 ksteps, 4 ntiles
        const int f = frag - NFRAG_W1, ks = f & 1, ln = f >> 1;
        src = W2 + (ln >> 2) * HID * HID; ld = HID; k0 = ks * 32; n0 = (ln & 3) * 16;
    } else {                                     // W3[l] (64x128): 2 ksteps, 8 ctiles
        const int f = frag - NFRAG_W1 - NFRAG_W2, ks = f & 1, lc = f >> 1;
        src = W3 + (lc >> 3) * HID * CH;  ld = CH;  k0 = ks * 32; n0 = (lc & 7) * 16;
    }
    // B layout: lane half h holds K=k0+8h+(0..7) in halves 0..7, K+16 in halves 8..15, at N=n0+lane.
    const int n  = n0 + (lane & 15);
    const int kb = k0 + ((lane >> 4) << 3);
    _Float16* dst = ws + ((size_t)frag * 32 + lane) * 16;
#pragma unroll
    for (int h = 0; h < 16; ++h) dst[h] = (_Float16)src[(kb + (h < 8 ? h : h + 8)) * ld + n];
}

// ---------------------------------- main --------------------------------------
__global__ __launch_bounds__(256)
void EdgeDegreeEmbeddingNetwork_higherorder_89979564851620_kernel(
    const int* __restrict__ atomic_numbers,         // (N,)
    const float* __restrict__ edge_vec,             // (N,K,3)
    const int* __restrict__ f_idx,                  // (N,K)
    const unsigned char* __restrict__ attn_mask,    // (N,K,1) bool
    const float* __restrict__ edge_scalars,         // (N,K,NB)
    const float* __restrict__ src_emb,              // (300,NB)
    const float* __restrict__ tgt_emb,              // (300,NB)
    const float* __restrict__ b1, const float* __restrict__ g1,
    const float* __restrict__ be1,
    const float* __restrict__ b2, const float* __restrict__ g2,
    const float* __restrict__ be2,
    const float* __restrict__ b3,
    const _Float16* __restrict__ wfrag,             // swizzled f16 weights
    float* __restrict__ out)                        // (N,9,CH)
{
    __shared__ alignas(16) _Float16 sX[KNB * DIN];        // 32x192 f16
    __shared__ alignas(16) _Float16 sH16[NL * KNB * HID]; // activations f16
    __shared__ float    sHraw[NL * KNB * HID];            // GEMM accum f32
    __shared__ float    sSh[KNB * 9];                     // masked sph. harmonics
    __shared__ float    sOut[9 * CH];                     // per-atom output acc

    const int n    = blockIdx.x;
    const int tid  = threadIdx.x;
    const int wave = tid >> 5;

    for (int i = tid; i < 9 * CH; i += 256) sOut[i] = 0.0f;

    // ---- Stage A: X = [edge_scalars | tgt_emb[z_n] | src_emb[z_src]] -> f16 ----
    {
        const int k = tid >> 3;            // edge 0..31
        const int chunk = tid & 7;         // 24 columns each
        const int zn   = atomic_numbers[n];
        const int zsrc = atomic_numbers[f_idx[n * KNB + k]];
        const float* es = edge_scalars + ((size_t)n * KNB + k) * NB;
        _Float16* xr = sX + k * DIN;
        const int j0 = chunk * 24;
#pragma unroll
        for (int j = j0; j < j0 + 24; ++j) {
            float v;
            if (j < NB)            v = es[j];
            else if (j < 2 * NB)   v = tgt_emb[zn * NB + (j - NB)];
            else                   v = src_emb[zsrc * NB + (j - 2 * NB)];
            xr[j] = (_Float16)v;
        }
    }
    // ---- spherical harmonics (l=0,1,2), attn_mask folded in ----
    if (tid < KNB) {
        const float* ev = edge_vec + ((size_t)n * KNB + tid) * 3;
        float x = ev[0], y = ev[1], z = ev[2];
        const float rn = rsqrtf(x * x + y * y + z * z);
        x *= rn; y *= rn; z *= rn;
        const float m = attn_mask[(size_t)n * KNB + tid] ? 0.0f : 1.0f;
        const float s3 = 1.7320508075688772f;
        float* sh = sSh + tid * 9;
        sh[0] = m;
        sh[1] = m * y;  sh[2] = m * z;  sh[3] = m * x;
        sh[4] = m * s3 * x * y;
        sh[5] = m * s3 * y * z;
        sh[6] = m * (1.5f * z * z - 0.5f);
        sh[7] = m * s3 * x * z;
        sh[8] = m * 0.5f * s3 * (x * x - y * y);
    }
    __syncthreads();

    // ---- GEMM1: X(32x192) @ W1[l](192x64); A-frags loaded once, reused 3 l's ----
    {
        const int mt = wave >> 2, nt = wave & 3;   // wave owns one (mt,nt) tile
        v16h a[6];
#pragma unroll
        for (int ks = 0; ks < 6; ++ks) a[ks] = load_A_lds(sX, mt * 16, ks * 32, DIN);
#pragma unroll
        for (int l = 0; l < NL; ++l) {
            v8f acc = {};
#pragma unroll
            for (int ks = 0; ks < 6; ++ks) {
                v16h b = load_B_ws(wfrag, (l * 4 + nt) * 6 + ks);
                acc = wmma_f16(a[ks], b, acc);
            }
            store_acc(sHraw + l * KNB * HID, mt * 16, nt * 16, HID, acc);
        }
    }
    __syncthreads();
    ln_silu(b1, g1, be1, sHraw, sH16);
    __syncthreads();

    // ---- GEMM2: H(32x64) @ W2[l](64x64) ----
    for (int t = wave; t < 24; t += 8) {
        const int l = t >> 3, r = t & 7, mt = r >> 2, nt = r & 3;
        v8f acc = {};
#pragma unroll
        for (int ks = 0; ks < 2; ++ks) {
            v16h a = load_A_lds(sH16 + l * KNB * HID, mt * 16, ks * 32, HID);
            v16h b = load_B_ws(wfrag + WS_W2_OFF, (l * 4 + nt) * 2 + ks);
            acc = wmma_f16(a, b, acc);
        }
        store_acc(sHraw + l * KNB * HID, mt * 16, nt * 16, HID, acc);
    }
    __syncthreads();
    ln_silu(b2, g2, be2, sHraw, sH16);
    __syncthreads();

    // ---- GEMM3: H(32x64) @ W3[l](64x128) + fused neighbor einsum ----
    for (int t = wave; t < 48; t += 8) {
        const int l = t >> 4, r = t & 15, mt = r >> 3, ct = r & 7;
        v8f acc = {};
#pragma unroll
        for (int ks = 0; ks < 2; ++ks) {
            v16h a = load_A_lds(sH16 + l * KNB * HID, mt * 16, ks * 32, HID);
            v16h b = load_B_ws(wfrag + WS_W3_OFF, (l * 8 + ct) * 2 + ks);
            acc = wmma_f16(a, b, acc);
        }
        if (l == 0)      reduce_einsum<1, 0>(acc, mt, ct, b3,          sSh, sOut);
        else if (l == 1) reduce_einsum<3, 1>(acc, mt, ct, b3 + CH,     sSh, sOut);
        else             reduce_einsum<5, 4>(acc, mt, ct, b3 + 2 * CH, sSh, sOut);
    }
    __syncthreads();

    // ---- store (N,9,CH) / AVG_AGGREGATE_NUM, non-temporal (write-once) ----
    float* op = out + (size_t)n * 9 * CH;
    for (int i = tid; i < 9 * CH; i += 256)
        __builtin_nontemporal_store(sOut[i] * 0.1f, op + i);
}

extern "C" void kernel_launch(void* const* d_in, const int* in_sizes, int n_in,
                              void* d_out, int out_size, void* d_ws, size_t ws_size,
                              hipStream_t stream) {
    (void)n_in; (void)ws_size; (void)out_size;
    const int* atomic_numbers        = (const int*)d_in[0];
    const float* edge_vec            = (const float*)d_in[1];
    const int* f_idx                 = (const int*)d_in[2];
    const unsigned char* attn_mask   = (const unsigned char*)d_in[3];
    const float* edge_scalars        = (const float*)d_in[4];
    const float* src_emb             = (const float*)d_in[5];
    const float* tgt_emb             = (const float*)d_in[6];
    const float* W1  = (const float*)d_in[7];
    const float* b1  = (const float*)d_in[8];
    const float* g1  = (const float*)d_in[9];
    const float* be1 = (const float*)d_in[10];
    const float* W2  = (const float*)d_in[11];
    const float* b2  = (const float*)d_in[12];
    const float* g2  = (const float*)d_in[13];
    const float* be2 = (const float*)d_in[14];
    const float* W3  = (const float*)d_in[15];
    const float* b3  = (const float*)d_in[16];
    float* out = (float*)d_out;
    _Float16* wfrag = (_Float16*)d_ws;   // needs 144*512*2 = 147456 bytes

    // 1) swizzle weights into f16 B-fragments (tiny; same stream -> ordered)
    edn_prep_weights_kernel<<<(NFRAG * 32 + 255) / 256, 256, 0, stream>>>(
        W1, W2, W3, wfrag);

    // 2) main fused kernel: one workgroup per atom
    const int N = in_sizes[0];
    EdgeDegreeEmbeddingNetwork_higherorder_89979564851620_kernel
        <<<N, 256, 0, stream>>>(atomic_numbers, edge_vec, f_idx, attn_mask,
                                edge_scalars, src_emb, tgt_emb,
                                b1, g1, be1, b2, g2, be2, b3, wfrag, out);
}
